// QwenDoubleStreamAttention_807453852384
// MI455X (gfx1250) — matrix operand order repaired and checked by reading, not compiled, not run on hardware
//
#include <hip/hip_runtime.h>
#include <hip/hip_bf16.h>
#include <math.h>

#define D_    3072
#define NH_   24
#define DH_   128
#define B_    2
#define SIMG  2048
#define STXT  256
#define STOT  2304
#define EPS_  1e-5f
#define SCALE_ 0.08838834764831843f  /* 128^-0.5 */

typedef __bf16 bf16;
typedef __attribute__((ext_vector_type(16))) __bf16 v16bf;
typedef __attribute__((ext_vector_type(8)))  __bf16 v8bf;
typedef __attribute__((ext_vector_type(4)))  __bf16 v4bf;
typedef __attribute__((ext_vector_type(8)))  float  v8f;

#define SHUF16(lo, hi) __builtin_shufflevector(lo, hi, 0,1,2,3,4,5,6,7,8,9,10,11,12,13,14,15)
#define WMMA_BF16(a, b, c) __builtin_amdgcn_wmma_f32_16x16x32_bf16(false, a, false, b, (short)0, c, false, false)

static __device__ __forceinline__ float hredmax(float x) {
  x = fmaxf(x, __shfl_xor(x, 1, 32));
  x = fmaxf(x, __shfl_xor(x, 2, 32));
  x = fmaxf(x, __shfl_xor(x, 4, 32));
  x = fmaxf(x, __shfl_xor(x, 8, 32));
  return x;
}
static __device__ __forceinline__ float hredsum(float x) {
  x += __shfl_xor(x, 1, 32);
  x += __shfl_xor(x, 2, 32);
  x += __shfl_xor(x, 4, 32);
  x += __shfl_xor(x, 8, 32);
  return x;
}

// ---------------- f32 -> bf16 convert (vector x4) ----------------
__global__ void k_cvt(const float* __restrict__ src, bf16* __restrict__ dst, int n4) {
  int i = blockIdx.x * blockDim.x + threadIdx.x;
  if (i >= n4) return;
  float4 v = ((const float4*)src)[i];
  v4bf o;
  o[0] = (bf16)v.x; o[1] = (bf16)v.y; o[2] = (bf16)v.z; o[3] = (bf16)v.w;
  ((v4bf*)dst)[i] = o;
}

// ---------------- weight f32 [K][N] -> bf16 transposed [N][K] ----------------
__global__ void k_wt(const float* __restrict__ W, bf16* __restrict__ Wt) {
  __shared__ float tile[32][33];
  int k0 = blockIdx.x * 32, n0 = blockIdx.y * 32;
  int tx = threadIdx.x & 31, ty = threadIdx.x >> 5;   // ty: 0..7
  for (int r = ty; r < 32; r += 8)
    tile[r][tx] = W[(size_t)(k0 + r) * D_ + n0 + tx];
  __syncthreads();
  for (int r = ty; r < 32; r += 8)
    Wt[(size_t)(n0 + r) * D_ + k0 + tx] = (bf16)tile[tx][r];
}

// ---------------- attention mask -> additive bias ----------------
__global__ void k_bias(const int* __restrict__ mask, float* __restrict__ bias) {
  int i = blockIdx.x * blockDim.x + threadIdx.x;
  if (i >= B_ * STOT) return;
  int b = i / STOT, s = i - b * STOT;
  float v = 0.f;
  if (s >= SIMG) v = mask[b * STXT + (s - SIMG)] ? 0.f : -1.0e9f;
  bias[i] = v;
}

// ---------------- QKV projection (+ optional RMSNorm + RoPE) ----------------
// Block = 8 waves covering 128(M) x 128(one head). Wave = 32(M) x 64(N):
// 2 A-frags x 4 B-frags -> 8 WMMAs per k-step, each B-frag reused twice.
// mode 1 (Q/K): out = [B][H][STOT][DH] bf16, RMSNorm + RoPE applied.
// mode 0 (V):   out = [B][H][DH][STOT] bf16 (transposed for PV B-fragments).
__global__ __launch_bounds__(256)
void k_qkv(const bf16* __restrict__ X, const bf16* __restrict__ Wt,
           const float* __restrict__ bias, const float* __restrict__ g,
           const float* __restrict__ fcos, const float* __restrict__ fsin,
           bf16* __restrict__ out, int s_len, int s_off, int mode)
{
  const int head = blockIdx.y;
  const int wave = threadIdx.x >> 5;
  const int lane = threadIdx.x & 31;
  const int lrow = lane & 15;
  const int lhi  = lane >> 4;
  const int rg   = wave >> 1;                 // row group 0..3 (32 rows each)
  const int cg   = wave & 1;                  // col group 0..1 (64 cols each)
  const int m0   = blockIdx.x * 128 + rg * 32;
  const int n0   = head * DH_ + cg * 64;

  __shared__ __align__(16) float Cs[128][DH_];   // 64 KB
  __shared__ float ps[128][2];
  __shared__ float rs[128];

  v8f acc[2][4];
  for (int rr = 0; rr < 2; ++rr)
    for (int t = 0; t < 4; ++t) acc[rr][t] = (v8f){0.f,0.f,0.f,0.f,0.f,0.f,0.f,0.f};

  const bf16* arow0 = X  + (size_t)(m0 + lrow) * D_ + lhi * 16;
  const bf16* arow1 = X  + (size_t)(m0 + 16 + lrow) * D_ + lhi * 16;
  const bf16* brow  = Wt + (size_t)(n0 + lrow) * D_ + lhi * 16;

  for (int k0 = 0; k0 < D_; k0 += 32) {
    v8bf a0lo = *(const v8bf*)(arow0 + k0);
    v8bf a0hi = *(const v8bf*)(arow0 + k0 + 8);
    v16bf a0  = SHUF16(a0lo, a0hi);
    v8bf a1lo = *(const v8bf*)(arow1 + k0);
    v8bf a1hi = *(const v8bf*)(arow1 + k0 + 8);
    v16bf a1  = SHUF16(a1lo, a1hi);
    __builtin_prefetch((const void*)(brow + k0 + 1024), 0, 3);
#pragma unroll
    for (int t = 0; t < 4; ++t) {
      v16bf bbf = *(const v16bf*)(brow + (size_t)t * 16 * D_ + k0);
      acc[0][t] = WMMA_BF16(a0, bbf, acc[0][t]);
      acc[1][t] = WMMA_BF16(a1, bbf, acc[1][t]);
    }
  }

  if (mode == 0) {  // V: bias + store transposed
#pragma unroll
    for (int t = 0; t < 4; ++t) {
      const float bc = bias[n0 + t * 16 + lrow];
      const int dh = cg * 64 + t * 16 + lrow;
      for (int rr = 0; rr < 2; ++rr)
        for (int i = 0; i < 8; ++i) {
          int mg = m0 + rr * 16 + lhi * 8 + i;
          int b  = mg / s_len;
          int s  = s_off + (mg - b * s_len);
          out[(((size_t)b * NH_ + head) * DH_ + dh) * STOT + s] = (bf16)(acc[rr][t][i] + bc);
        }
    }
    return;
  }

  // Q/K: stash 128x128 block in LDS for row-wise RMSNorm
#pragma unroll
  for (int t = 0; t < 4; ++t) {
    const float bc = bias[n0 + t * 16 + lrow];
    for (int rr = 0; rr < 2; ++rr)
      for (int i = 0; i < 8; ++i)
        Cs[rg * 32 + rr * 16 + lhi * 8 + i][cg * 64 + t * 16 + lrow] = acc[rr][t][i] + bc;
  }
  __syncthreads();

  {  // partial sums of squares: 2 threads per row
    int r = threadIdx.x >> 1, q = threadIdx.x & 1;
    float ssum = 0.f;
    for (int c = q * 64; c < q * 64 + 64; ++c) { float v = Cs[r][c]; ssum += v * v; }
    ps[r][q] = ssum;
  }
  __syncthreads();
  if (threadIdx.x < 128) {
    float ssum = ps[threadIdx.x][0] + ps[threadIdx.x][1];
    rs[threadIdx.x] = rsqrtf(ssum * (1.0f / DH_) + EPS_);
  }
  __syncthreads();

  {  // apply gain + RoPE, store bf16 (each thread: one row, 64 cols)
    int r  = threadIdx.x >> 1;
    int c0 = (threadIdx.x & 1) * 64;
    int mg = blockIdx.x * 128 + r;
    int b  = mg / s_len;
    int sl = mg - b * s_len;
    int s  = s_off + sl;
    float sc = rs[r];
    bf16* orow = out + (((size_t)b * NH_ + head) * STOT + s) * DH_;
    for (int jj = 0; jj < 8; ++jj) {
      int cb = c0 + jj * 8;
      v8bf ov;
      for (int j = 0; j < 4; ++j) {
        int c = cb + 2 * j;
        float x0 = Cs[r][c]     * sc * g[c];
        float x1 = Cs[r][c + 1] * sc * g[c + 1];
        float co = fcos[(size_t)sl * (DH_ / 2) + (c >> 1)];
        float si = fsin[(size_t)sl * (DH_ / 2) + (c >> 1)];
        ov[2 * j]     = (bf16)(x0 * co - x1 * si);
        ov[2 * j + 1] = (bf16)(x0 * si + x1 * co);
      }
      *(v8bf*)(orow + cb) = ov;
    }
  }
}

// ---------------- flash attention with async-to-LDS K/V staging ----------------
// 8 waves share one (b, head): K-tile (32x128) and V-tile (128x32) staged
// cooperatively with global_load_async_to_lds_b128, double-buffered so the
// next tile's DMA overlaps the current tile's 16 WMMAs per wave.
__global__ __launch_bounds__(256)
void k_attn(const bf16* __restrict__ Q, const bf16* __restrict__ K,
            const bf16* __restrict__ Vt, const float* __restrict__ biasbuf,
            bf16* __restrict__ O)
{
  const int wave = threadIdx.x >> 5;
  const int lane = threadIdx.x & 31;
  const int lrow = lane & 15;
  const int lhi  = lane >> 4;
  const int q0   = (blockIdx.x * 8 + wave) * 16;
  const int head = blockIdx.y;
  const int b    = blockIdx.z;

  __shared__ __align__(16) bf16 Ks[2][32][DH_];   // 2 x 8 KB
  __shared__ __align__(16) bf16 Vs[2][DH_][32];   // 2 x 8 KB
  __shared__ __align__(16) bf16 Ps[8][16][32];    // 8 KB

  const size_t bh = (size_t)b * NH_ + head;
  const bf16* Qb = Q  + bh * STOT * DH_;
  const bf16* Kb = K  + bh * STOT * DH_;
  const bf16* Vb = Vt + bh * DH_ * STOT;
  const float* bbp = biasbuf + (size_t)b * STOT;

  // cooperative async stage of one 32-key step: 512 x 16B chunks each for K and V
  auto stage = [&](int buf, int kt) {
#pragma unroll
    for (int p = 0; p < 2; ++p) {
      int ch = threadIdx.x + p * 256;          // 0..511
      int kr = ch >> 4, kc = (ch & 15) * 8;    // K: 16 chunks per 256B row
      uint32_t ldsK = (uint32_t)(uintptr_t)(&Ks[buf][kr][kc]);
      const bf16* gK = Kb + (size_t)(kt + kr) * DH_ + kc;
      asm volatile("global_load_async_to_lds_b128 %0, %1, off"
                   :: "v"(ldsK), "v"(gK) : "memory");
      int vr = ch >> 2, vc = (ch & 3) * 8;     // V: 4 chunks per 64B row
      uint32_t ldsV = (uint32_t)(uintptr_t)(&Vs[buf][vr][vc]);
      const bf16* gV = Vb + (size_t)vr * STOT + kt + vc;
      asm volatile("global_load_async_to_lds_b128 %0, %1, off"
                   :: "v"(ldsV), "v"(gV) : "memory");
    }
  };

  v16bf qf[4];
  for (int c = 0; c < 4; ++c) {
    const bf16* qp = Qb + (size_t)(q0 + lrow) * DH_ + c * 32;
    v8bf lo = *(const v8bf*)(qp + lhi * 16);
    v8bf hi = *(const v8bf*)(qp + 8 + lhi * 16);
    qf[c] = SHUF16(lo, hi);
  }

  v8f o[8];
  v8f mrun, lrun;
  for (int t = 0; t < 8; ++t)
    o[t] = (v8f){0.f,0.f,0.f,0.f,0.f,0.f,0.f,0.f};
  for (int i = 0; i < 8; ++i) { mrun[i] = -3.0e38f; lrun[i] = 0.f; }

  stage(0, 0);
  for (int it = 0, kt = 0; kt < STOT; ++it, kt += 32) {
    const int cur = it & 1;
    const bool have_next = (kt + 32) < STOT;
    if (have_next) {
      stage(cur ^ 1, kt + 32);
      asm volatile("s_wait_asynccnt 0x4" ::: "memory");  // current tile's 4 loads done
    } else {
      asm volatile("s_wait_asynccnt 0x0" ::: "memory");
    }
    __syncthreads();

    v8f s0 = {0.f,0.f,0.f,0.f,0.f,0.f,0.f,0.f};
    v8f s1 = {0.f,0.f,0.f,0.f,0.f,0.f,0.f,0.f};
#pragma unroll
    for (int c = 0; c < 4; ++c) {
      v16bf bk0 = *(const v16bf*)(&Ks[cur][lrow][c * 32 + lhi * 16]);
      s0 = WMMA_BF16(qf[c], bk0, s0);
      v16bf bk1 = *(const v16bf*)(&Ks[cur][16 + lrow][c * 32 + lhi * 16]);
      s1 = WMMA_BF16(qf[c], bk1, s1);
    }
    const float bias0 = bbp[kt + lrow];
    const float bias1 = bbp[kt + 16 + lrow];
    v8f mnew, corr, p0, p1;
    for (int i = 0; i < 8; ++i) {
      s0[i] = s0[i] * SCALE_ + bias0;
      s1[i] = s1[i] * SCALE_ + bias1;
      float t = hredmax(fmaxf(s0[i], s1[i]));
      mnew[i] = fmaxf(mrun[i], t);
    }
    for (int i = 0; i < 8; ++i) {
      corr[i] = __expf(mrun[i] - mnew[i]);
      p0[i] = __expf(s0[i] - mnew[i]);
      p1[i] = __expf(s1[i] - mnew[i]);
      lrun[i] = lrun[i] * corr[i] + hredsum(p0[i] + p1[i]);
    }
    mrun = mnew;
#pragma unroll
    for (int t = 0; t < 8; ++t) o[t] *= corr;

    // C-layout probs -> LDS -> A-layout bf16 fragment (wave-private region)
    for (int i = 0; i < 8; ++i) {
      Ps[wave][lhi * 8 + i][lrow]      = (bf16)p0[i];
      Ps[wave][lhi * 8 + i][16 + lrow] = (bf16)p1[i];
    }
    v8bf plo = *(const v8bf*)(&Ps[wave][lrow][lhi * 16]);
    v8bf phi = *(const v8bf*)(&Ps[wave][lrow][8 + lhi * 16]);
    v16bf pa = SHUF16(plo, phi);

#pragma unroll
    for (int t = 0; t < 8; ++t) {
      v16bf bv = *(const v16bf*)(&Vs[cur][t * 16 + lrow][lhi * 16]);
      o[t] = WMMA_BF16(pa, bv, o[t]);
    }
    __syncthreads();  // done reading 'cur' before it is restaged next iter
  }

  for (int i = 0; i < 8; ++i) lrun[i] = 1.0f / lrun[i];
  for (int t = 0; t < 8; ++t)
    for (int i = 0; i < 8; ++i)
      O[((size_t)b * STOT + q0 + lhi * 8 + i) * D_ + head * DH_ + t * 16 + lrow] =
          (bf16)(o[t][i] * lrun[i]);
}

// ---------------- output projection GEMM -> f32 ----------------
// Block 128x128, wave 32x64, no LDS in K loop.
__global__ __launch_bounds__(256)
void k_out(const bf16* __restrict__ A, const bf16* __restrict__ Wt,
           const float* __restrict__ bias, float* __restrict__ out, int M)
{
  const int wave = threadIdx.x >> 5;
  const int lane = threadIdx.x & 31;
  const int lrow = lane & 15;
  const int lhi  = lane >> 4;
  const int rg   = wave >> 1;
  const int cg   = wave & 1;
  const int m0   = blockIdx.x * 128 + rg * 32;
  const int n0   = blockIdx.y * 128 + cg * 64;

  v8f acc[2][4];
  for (int rr = 0; rr < 2; ++rr)
    for (int t = 0; t < 4; ++t) acc[rr][t] = (v8f){0.f,0.f,0.f,0.f,0.f,0.f,0.f,0.f};

  const bf16* arow0 = A  + (size_t)(m0 + lrow) * D_ + lhi * 16;
  const bf16* arow1 = A  + (size_t)(m0 + 16 + lrow) * D_ + lhi * 16;
  const bf16* brow  = Wt + (size_t)(n0 + lrow) * D_ + lhi * 16;

  for (int k0 = 0; k0 < D_; k0 += 32) {
    v8bf a0lo = *(const v8bf*)(arow0 + k0);
    v8bf a0hi = *(const v8bf*)(arow0 + k0 + 8);
    v16bf a0  = SHUF16(a0lo, a0hi);
    v8bf a1lo = *(const v8bf*)(arow1 + k0);
    v8bf a1hi = *(const v8bf*)(arow1 + k0 + 8);
    v16bf a1  = SHUF16(a1lo, a1hi);
    __builtin_prefetch((const void*)(brow + k0 + 1024), 0, 3);
#pragma unroll
    for (int t = 0; t < 4; ++t) {
      v16bf bbf = *(const v16bf*)(brow + (size_t)t * 16 * D_ + k0);
      acc[0][t] = WMMA_BF16(a0, bbf, acc[0][t]);
      acc[1][t] = WMMA_BF16(a1, bbf, acc[1][t]);
    }
  }

#pragma unroll
  for (int t = 0; t < 4; ++t) {
    const float bc = bias[n0 + t * 16 + lrow];
    for (int rr = 0; rr < 2; ++rr)
      for (int i = 0; i < 8; ++i)
        out[(size_t)(m0 + rr * 16 + lhi * 8 + i) * D_ + n0 + t * 16 + lrow] = acc[rr][t][i] + bc;
  }
}

// =====================================================================
extern "C" void kernel_launch(void* const* d_in, const int* in_sizes, int n_in,
                              void* d_out, int out_size, void* d_ws, size_t ws_size,
                              hipStream_t stream)
{
  (void)in_sizes; (void)n_in; (void)out_size; (void)ws_size;

  const float* hidden = (const float*)d_in[0];
  const float* enc    = (const float*)d_in[1];
  const int*   mask   = (const int*)d_in[2];
  const float* icos   = (const float*)d_in[4];
  const float* isin   = (const float*)d_in[5];
  const float* tcos   = (const float*)d_in[6];
  const float* tsin   = (const float*)d_in[7];
  const float* w[8]   = { (const float*)d_in[8],  (const float*)d_in[10],
                          (const float*)d_in[12], (const float*)d_in[14],
                          (const float*)d_in[16], (const float*)d_in[18],
                          (const float*)d_in[20], (const float*)d_in[22] };
  const float* bq     = (const float*)d_in[9];
  const float* bk     = (const float*)d_in[11];
  const float* bv     = (const float*)d_in[13];
  const float* bq_add = (const float*)d_in[15];
  const float* bk_add = (const float*)d_in[17];
  const float* bv_add = (const float*)d_in[19];
  const float* bo     = (const float*)d_in[21];
  const float* bo_add = (const float*)d_in[23];
  const float* gq     = (const float*)d_in[24];
  const float* gk     = (const float*)d_in[25];
  const float* gq_add = (const float*)d_in[26];
  const float* gk_add = (const float*)d_in[27];

  char* ws = (char*)d_ws;
  auto alloc = [&](size_t bytes) -> char* {
    char* p = ws;
    ws += (bytes + 255) & ~(size_t)255;
    return p;
  };

  bf16* Xh = (bf16*)alloc((size_t)B_ * SIMG * D_ * 2);
  bf16* Xe = (bf16*)alloc((size_t)B_ * STXT * D_ * 2);
  bf16* Wt[8];
  for (int i = 0; i < 8; ++i) Wt[i] = (bf16*)alloc((size_t)D_ * D_ * 2);
  bf16* Qb  = (bf16*)alloc((size_t)B_ * NH_ * STOT * DH_ * 2);
  bf16* Kb  = (bf16*)alloc((size_t)B_ * NH_ * STOT * DH_ * 2);
  bf16* Vtb = (bf16*)alloc((size_t)B_ * NH_ * STOT * DH_ * 2);
  bf16* Ob  = (bf16*)alloc((size_t)B_ * STOT * D_ * 2);
  float* bb = (float*)alloc((size_t)B_ * STOT * 4);

  { int n4 = B_ * SIMG * D_ / 4; k_cvt<<<(n4 + 255) / 256, 256, 0, stream>>>(hidden, Xh, n4); }
  { int n4 = B_ * STXT * D_ / 4; k_cvt<<<(n4 + 255) / 256, 256, 0, stream>>>(enc, Xe, n4); }

  dim3 wtg(D_ / 32, D_ / 32);
  for (int i = 0; i < 8; ++i) k_wt<<<wtg, 256, 0, stream>>>(w[i], Wt[i]);

  k_bias<<<(B_ * STOT + 255) / 256, 256, 0, stream>>>(mask, bb);

  dim3 gi(B_ * SIMG / 128, NH_);
  dim3 gt(B_ * STXT / 128, NH_);
  k_qkv<<<gi, 256, 0, stream>>>(Xh, Wt[0], bq, gq, icos, isin, Qb,  SIMG, 0, 1);
  k_qkv<<<gi, 256, 0, stream>>>(Xh, Wt[1], bk, gk, icos, isin, Kb,  SIMG, 0, 1);
  k_qkv<<<gi, 256, 0, stream>>>(Xh, Wt[2], bv, nullptr, nullptr, nullptr, Vtb, SIMG, 0, 0);
  k_qkv<<<gt, 256, 0, stream>>>(Xe, Wt[3], bq_add, gq_add, tcos, tsin, Qb,  STXT, SIMG, 1);
  k_qkv<<<gt, 256, 0, stream>>>(Xe, Wt[4], bk_add, gk_add, tcos, tsin, Kb,  STXT, SIMG, 1);
  k_qkv<<<gt, 256, 0, stream>>>(Xe, Wt[5], bv_add, nullptr, nullptr, nullptr, Vtb, STXT, SIMG, 0);

  dim3 ga(STOT / 128, NH_, B_);
  k_attn<<<ga, 256, 0, stream>>>(Qb, Kb, Vtb, bb, Ob);

  float* out = (float*)d_out;
  dim3 go1(SIMG / 128, D_ / 128);
  dim3 go2(STXT / 128, D_ / 128);
  k_out<<<go1, 256, 0, stream>>>(Ob,                              Wt[6], bo,     out,                                             SIMG);
  k_out<<<go1, 256, 0, stream>>>(Ob + (size_t)STOT * D_,          Wt[6], bo,     out + (size_t)SIMG * D_,                         SIMG);
  k_out<<<go2, 256, 0, stream>>>(Ob + (size_t)SIMG * D_,          Wt[7], bo_add, out + (size_t)2 * SIMG * D_,                     STXT);
  k_out<<<go2, 256, 0, stream>>>(Ob + ((size_t)STOT + SIMG) * D_, Wt[7], bo_add, out + (size_t)2 * SIMG * D_ + (size_t)STXT * D_, STXT);
}
